// MultipleNegativesRankingLoss_63874753626759
// MI455X (gfx1250) — matrix operand order repaired and checked by reading, not compiled, not run on hardware
//
#include <hip/hip_runtime.h>
#include <hip/hip_bf16.h>
#include <math.h>

typedef __attribute__((ext_vector_type(16))) __bf16 v16bf;
typedef __attribute__((ext_vector_type(8)))  __bf16 v8bf;
typedef __attribute__((ext_vector_type(4)))  __bf16 v4bf;
typedef __attribute__((ext_vector_type(8)))  float  v8f;
typedef __attribute__((ext_vector_type(4)))  float  v4f;

#define B_ROWS 4096
#define C_COLS 8192   // pos (4096) + neg (4096)
#define DIM    1024
#define LOSS_SCALE 20.0f

// ---------------------------------------------------------------------------
// Kernel 0: zero the per-row sum-of-exp accumulators (ws is poisoned 0xAA)
// ---------------------------------------------------------------------------
__global__ void init_rowsum_kernel(float* __restrict__ rowsum) {
    int i = blockIdx.x * 256 + threadIdx.x;
    if (i < B_ROWS) rowsum[i] = 0.0f;
}

// ---------------------------------------------------------------------------
// Kernel 1: L2-normalize each row and cast to bf16.
// One 256-thread block per row (4 floats/thread over D=1024).
// Rows [0, 4096) -> q; rows [4096, 8192) -> pos; rows [8192, 12288) -> neg.
// ---------------------------------------------------------------------------
__global__ __launch_bounds__(256) void normalize_cast_kernel(
    const float* __restrict__ q, const float* __restrict__ pos,
    const float* __restrict__ neg,
    __bf16* __restrict__ qn, __bf16* __restrict__ dn)
{
    const int row = blockIdx.x;
    const float* src;
    __bf16* dst;
    if (row < B_ROWS) {
        src = q + (size_t)row * DIM;
        dst = qn + (size_t)row * DIM;
    } else {
        int rd = row - B_ROWS;
        src = (rd < B_ROWS) ? pos + (size_t)rd * DIM
                            : neg + (size_t)(rd - B_ROWS) * DIM;
        dst = dn + (size_t)rd * DIM;
    }
    const int t = threadIdx.x;
    v4f x = *(const v4f*)(src + t * 4);
    float ss = x.x * x.x + x.y * x.y + x.z * x.z + x.w * x.w;
    // wave32 butterfly then cross-wave LDS reduce (8 waves / block)
    #pragma unroll
    for (int m = 16; m >= 1; m >>= 1) ss += __shfl_xor(ss, m, 32);
    __shared__ float sred[8];
    const int wave = t >> 5, lane = t & 31;
    if (lane == 0) sred[wave] = ss;
    __syncthreads();
    float tot = 0.0f;
    #pragma unroll
    for (int i = 0; i < 8; ++i) tot += sred[i];
    const float inv = 1.0f / fmaxf(sqrtf(tot), 1e-8f);  // x / max(||x||, eps)
    v4bf o;
    o.x = (__bf16)(x.x * inv);
    o.y = (__bf16)(x.y * inv);
    o.z = (__bf16)(x.z * inv);
    o.w = (__bf16)(x.w * inv);
    *(v4bf*)(dst + t * 4) = o;
}

// ---------------------------------------------------------------------------
// Kernel 2: diagonal scores s_bb = 20 * dot(qn[b], dn[b]).
// One 256-thread block per row; trivial vs the GEMM (8 MFLOP total) but
// removes all exec-mask-churning diag checks from the GEMM epilogue.
// ---------------------------------------------------------------------------
__global__ __launch_bounds__(256) void diag_kernel(
    const __bf16* __restrict__ qn, const __bf16* __restrict__ dn,
    float* __restrict__ diag)
{
    const int row = blockIdx.x;
    const int t = threadIdx.x;
    v4bf a = *(const v4bf*)(qn + (size_t)row * DIM + t * 4);
    v4bf b = *(const v4bf*)(dn + (size_t)row * DIM + t * 4);
    float s = (float)a.x * (float)b.x + (float)a.y * (float)b.y +
              (float)a.z * (float)b.z + (float)a.w * (float)b.w;
    #pragma unroll
    for (int m = 16; m >= 1; m >>= 1) s += __shfl_xor(s, m, 32);
    __shared__ float sred[8];
    const int wave = t >> 5, lane = t & 31;
    if (lane == 0) sred[wave] = s;
    __syncthreads();
    if (t == 0) {
        float tot = 0.0f;
        #pragma unroll
        for (int i = 0; i < 8; ++i) tot += sred[i];
        diag[row] = LOSS_SCALE * tot;
    }
}

// ---------------------------------------------------------------------------
// Kernel 3: fused bf16 WMMA GEMM (scores = 20 * qn @ dn^T) + exp + row sums.
// Block = 128 threads = 4 waves arranged 2x2; each wave owns a 64x64 output
// tile (4x4 fragments of v_wmma_f32_16x16x32_bf16). K loop over D=1024 in
// steps of 32. Scores are bounded by |20|, so sum-of-exp is fp32-safe with
// no max subtraction -> single pass. Branch-free epilogue.
// ---------------------------------------------------------------------------
__global__ __launch_bounds__(128) void gemm_logsumexp_kernel(
    const __bf16* __restrict__ qn, const __bf16* __restrict__ dn,
    float* __restrict__ rowsum)
{
    const int lane  = threadIdx.x & 31;
    const int wave  = threadIdx.x >> 5;
    const int waveM = wave & 1;
    const int waveN = wave >> 1;
    const int rowBase = blockIdx.x * 128 + waveM * 64;
    const int colBase = blockIdx.y * 128 + waveN * 64;
    const int mr = lane & 15;   // row-in-subtile for A, col-in-subtile for B
    const int g  = lane >> 4;   // half-wave selector

    // Per-lane base pointers (this code's contiguous k order, same for A and B):
    // A lane L: row M=L%16, elems K = g*16 .. g*16+15 (contiguous)
    // B lane L: doc row N=L%16, elems K = g*16 .. g*16+15 (contiguous)
    const __bf16* aPtr[4];
    const __bf16* bPtr[4];
    #pragma unroll
    for (int i = 0; i < 4; ++i) {
        aPtr[i] = qn + (size_t)(rowBase + i * 16 + mr) * DIM + g * 16;
        bPtr[i] = dn + (size_t)(colBase + i * 16 + mr) * DIM + g * 16;
    }

    v8f zeroAcc = {};
    v8f acc[4][4];
    #pragma unroll
    for (int i = 0; i < 4; ++i)
        #pragma unroll
        for (int j = 0; j < 4; ++j)
            acc[i][j] = zeroAcc;

    for (int k0 = 0; k0 < DIM; k0 += 32) {
        v16bf a[4], b[4];
        #pragma unroll
        for (int i = 0; i < 4; ++i) {
            v8bf lo = *(const v8bf*)(aPtr[i] + k0);        // K = g*16 + 0..7
            v8bf hi = *(const v8bf*)(aPtr[i] + k0 + 8);    // K = g*16 + 8..15
            a[i] = __builtin_shufflevector(lo, hi,
                    0, 1, 2, 3, 4, 5, 6, 7, 8, 9, 10, 11, 12, 13, 14, 15);
            b[i] = *(const v16bf*)(bPtr[i] + k0);          // K = g*16 + 0..15
        }
        #pragma unroll
        for (int i = 0; i < 4; ++i)
            #pragma unroll
            for (int j = 0; j < 4; ++j)
                acc[i][j] = __builtin_amdgcn_wmma_f32_16x16x32_bf16(
                    /*neg_a=*/false, a[i], /*neg_b=*/false, b[j],
                    /*c_mod=*/(short)0, acc[i][j],
                    /*reuse_a=*/false, /*reuse_b=*/false);
    }

    // Epilogue. C/D frag layout: VGPR v -> M = v + 8*(lane/16), N = lane%16.
    // Per-lane partial sums over the wave's 64 columns, then reduce the 16
    // lanes of each half-wave, then one atomic per row from lanes 0 and 16.
    const int halfSel = lane >> 4;
    const int nIdx    = lane & 15;
    #pragma unroll
    for (int i = 0; i < 4; ++i) {
        float rs[8];
        #pragma unroll
        for (int v = 0; v < 8; ++v) rs[v] = 0.0f;
        #pragma unroll
        for (int j = 0; j < 4; ++j)
            #pragma unroll
            for (int v = 0; v < 8; ++v)
                rs[v] += __expf(LOSS_SCALE * acc[i][j][v]);  // <= exp(20): safe
        #pragma unroll
        for (int v = 0; v < 8; ++v) {
            float s = rs[v];
            s += __shfl_xor(s, 1, 32);
            s += __shfl_xor(s, 2, 32);
            s += __shfl_xor(s, 4, 32);
            s += __shfl_xor(s, 8, 32);
            if (nIdx == 0) {  // lanes 0 (rows v) and 16 (rows 8+v)
                int rowg = rowBase + i * 16 + v + 8 * halfSel;
                atomicAdd(&rowsum[rowg], s);
            }
        }
    }
}

// ---------------------------------------------------------------------------
// Kernel 4: loss = mean_b( log(sum_exp_b) - score_bb )
// ---------------------------------------------------------------------------
__global__ __launch_bounds__(256) void loss_reduce_kernel(
    const float* __restrict__ rowsum, const float* __restrict__ diag,
    float* __restrict__ out)
{
    float s = 0.0f;
    for (int i = threadIdx.x; i < B_ROWS; i += 256)
        s += __logf(rowsum[i]) - diag[i];
    #pragma unroll
    for (int m = 16; m >= 1; m >>= 1) s += __shfl_xor(s, m, 32);
    __shared__ float sred[8];
    const int wave = threadIdx.x >> 5, lane = threadIdx.x & 31;
    if (lane == 0) sred[wave] = s;
    __syncthreads();
    if (threadIdx.x == 0) {
        float t = 0.0f;
        #pragma unroll
        for (int i = 0; i < 8; ++i) t += sred[i];
        out[0] = t / (float)B_ROWS;
    }
}

// ---------------------------------------------------------------------------
extern "C" void kernel_launch(void* const* d_in, const int* in_sizes, int n_in,
                              void* d_out, int out_size, void* d_ws, size_t ws_size,
                              hipStream_t stream) {
    const float* q   = (const float*)d_in[0];
    const float* pos = (const float*)d_in[1];
    const float* neg = (const float*)d_in[2];

    char* ws = (char*)d_ws;
    __bf16* qn     = (__bf16*)ws;                                   // 8 MB
    __bf16* dn     = (__bf16*)(ws + (size_t)B_ROWS * DIM * 2);      // 16 MB
    float*  rowsum = (float*)(ws + (size_t)(B_ROWS + C_COLS) * DIM * 2);
    float*  diag   = rowsum + B_ROWS;

    init_rowsum_kernel<<<(B_ROWS + 255) / 256, 256, 0, stream>>>(rowsum);
    normalize_cast_kernel<<<B_ROWS + C_COLS, 256, 0, stream>>>(q, pos, neg, qn, dn);
    diag_kernel<<<B_ROWS, 256, 0, stream>>>(qn, dn, diag);
    dim3 grid(B_ROWS / 128, C_COLS / 128);   // 32 x 64 workgroups
    gemm_logsumexp_kernel<<<grid, 128, 0, stream>>>(qn, dn, rowsum);
    loss_reduce_kernel<<<1, 256, 0, stream>>>(rowsum, diag, (float*)d_out);
}
